// JambaMambaDecoderLayer_82214263980219
// MI455X (gfx1250) — matrix operand order repaired and checked by reading, not compiled, not run on hardware
//
#include <hip/hip_runtime.h>
#include <hip/hip_bf16.h>

// ---------------------------------------------------------------------------
// MI455X (gfx1250) Jamba layer: wave32, WMMA 16x16x32 f16->f32,
// async global->LDS triple-buffered GEMM (ASYNCcnt pipelined, depth 2).
// Sizes: B=1 L=1024 D=2048 Di=4096 N=16 K=4 R=128 I=5632 E=8 top2.
// ---------------------------------------------------------------------------

typedef __attribute__((ext_vector_type(16))) _Float16 v16h;
typedef __attribute__((ext_vector_type(8)))  _Float16 v8h;
typedef __attribute__((ext_vector_type(8)))  float    v8f;

union Frag16 { v16h v; v8h h2[2]; };

static constexpr int T_  = 1024;   // tokens (B*L)
static constexpr int D_  = 2048;
static constexpr int Di_ = 4096;
static constexpr int N_  = 16;
static constexpr int R_  = 128;
static constexpr int I_  = 5632;
static constexpr int E_  = 8;

__device__ __forceinline__ float siluf(float x) { return x / (1.f + __expf(-x)); }
__device__ __forceinline__ float softplusf(float x) { return x > 20.f ? x : log1pf(__expf(x)); }

// per-lane async 16B global -> LDS copies (gfx1250, ASYNCcnt-tracked)
__device__ __forceinline__ void async_g2l_b128(unsigned lds_off, const float* gaddr) {
  asm volatile("global_load_async_to_lds_b128 %0, %1, off"
               :: "v"(lds_off), "v"(gaddr) : "memory");
}
// non-temporal flavor for once-streamed weight tiles (don't pollute L2)
__device__ __forceinline__ void async_g2l_b128_nt(unsigned lds_off, const float* gaddr) {
  asm volatile("global_load_async_to_lds_b128 %0, %1, off th:TH_LOAD_NT"
               :: "v"(lds_off), "v"(gaddr) : "memory");
}
__device__ __forceinline__ void wait_async0() {
  asm volatile("s_wait_asynccnt 0x0" ::: "memory");
}
__device__ __forceinline__ void wait_async8() {   // one 8-instruction tile may remain
  asm volatile("s_wait_asynccnt 0x8" ::: "memory");
}

// 8 consecutive LDS fp32 -> v8h (two aligned b128 loads + packed cvt;
// VALU cvts co-execute with XDL WMMAs)
__device__ __forceinline__ v8h cvt8(const float* p) {
  float4 x = *(const float4*)p;
  float4 y = *(const float4*)(p + 4);
  v8h r;
  r[0] = (_Float16)x.x; r[1] = (_Float16)x.y; r[2] = (_Float16)x.z; r[3] = (_Float16)x.w;
  r[4] = (_Float16)y.x; r[5] = (_Float16)y.y; r[6] = (_Float16)y.z; r[7] = (_Float16)y.w;
  return r;
}

// ---------------------------------------------------------------------------
// Generic WMMA GEMM:  C[m, n] = epi( sum_k A[rowA(m), k] * W[e][n, k] )
// Block tile 128(M) x 128(N), K-step 32; 8 waves, each owns a 32x64 sub-tile
// (2 A-frags x 4 B-frags = 8 WMMAs per K-step). Triple-buffered LDS fed by
// global_load_async_to_lds_b128 with prefetch depth 2; one barrier per K-step;
// the final K-step is peeled so the hot loop always waits asynccnt<=8.
// Out-of-range rows/cols are CLAMPED (masked at store): staging is branch-free.
// grid.z = expert (W advances by N*K, gather/scatter maps by mapStride).
// ---------------------------------------------------------------------------
#define BM 128
#define BN 128
#define BK 32
#define ASTR 36                      // padded LDS row stride (floats)
#define NBUF 3

__global__ __launch_bounds__(256) void gemm_wmma(
    const float* __restrict__ A, const float* __restrict__ W, float* __restrict__ C,
    int M, int N, int K, int lda, int ldc,
    const int* __restrict__ mapA, const int* __restrict__ mapC,
    const int* __restrict__ cnt, int mapStride,
    const float* __restrict__ bias, int epi)
{
  __shared__ alignas(16) float Ash[NBUF][BM * ASTR];
  __shared__ alignas(16) float Bsh[NBUF][BN * ASTR];

  const int e = blockIdx.z;
  const int Md = cnt ? cnt[e] : M;
  const int tile_m = blockIdx.y * BM;
  const int tile_n = blockIdx.x * BN;
  if (tile_m >= Md) return;                 // uniform across block

  const size_t Wbase = (size_t)e * (size_t)N * (size_t)K;
  const int* mA = mapA ? (mapA + e * mapStride) : nullptr;
  const int* mC = mapC ? (mapC + e * mapStride) : nullptr;

  const int tid  = threadIdx.x;
  const int lane = tid & 31;
  const int wave = tid >> 5;
  const int wm = wave >> 1;                 // 0..3 -> rows wm*32
  const int wn = wave & 1;                  // 0..1 -> cols wn*64
  const int ml = lane & 15;
  const int hi = lane >> 4;

  // ---- hoisted, clamped staging addresses; pointers advance per issue ----
  const float* ap[4]; unsigned aoff[4];
  const float* wp[4]; unsigned boff[4];
  const unsigned ashBase = (unsigned)(uintptr_t)&Ash[0][0];
  const unsigned bshBase = (unsigned)(uintptr_t)&Bsh[0][0];
#pragma unroll
  for (int i = 0; i < 4; ++i) {
    int eidx = tid + i * 256;               // 1024 b128 slots of the A tile
    int row  = eidx >> 3;
    int c4   = (eidx & 7) << 2;
    int gr   = tile_m + row;
    int grc  = gr < Md ? gr : (Md - 1);     // clamp: garbage rows masked later
    int ar   = mA ? mA[grc] : grc;
    ap[i]    = A + (size_t)ar * lda + c4;
    aoff[i]  = ashBase + (unsigned)(row * ASTR + c4) * 4u;
  }
#pragma unroll
  for (int i = 0; i < 4; ++i) {
    int eidx = tid + i * 256;               // 1024 b128 slots of the W tile
    int row  = eidx >> 3;
    int c4   = (eidx & 7) << 2;
    int gn   = tile_n + row;
    int gnc  = gn < N ? gn : (N - 1);       // clamp: garbage cols masked later
    wp[i]    = W + Wbase + (size_t)gnc * K + c4;
    boff[i]  = bshBase + (unsigned)(row * ASTR + c4) * 4u;
  }
  const unsigned ABUF = (unsigned)(BM * ASTR) * 4u;
  const unsigned BBUF = (unsigned)(BN * ASTR) * 4u;

  int ibuf = 0;                             // rotating issue-buffer index
  auto issue_tile = [&]() {                 // issues next K tile, advances state
#pragma unroll
    for (int i = 0; i < 4; ++i) {
      async_g2l_b128(aoff[i] + (unsigned)ibuf * ABUF, ap[i]);
      ap[i] += BK;
    }
#pragma unroll
    for (int i = 0; i < 4; ++i) {
      async_g2l_b128_nt(boff[i] + (unsigned)ibuf * BBUF, wp[i]);
      wp[i] += BK;
    }
    ibuf = (ibuf + 1 == NBUF) ? 0 : ibuf + 1;
  };

  const v8f vzero = {0.f, 0.f, 0.f, 0.f, 0.f, 0.f, 0.f, 0.f};
  v8f acc[2][4];
#pragma unroll
  for (int i = 0; i < 2; ++i)
#pragma unroll
    for (int j = 0; j < 4; ++j) acc[i][j] = vzero;

  auto compute = [&](int cb) {
    const float* As = Ash[cb];
    const float* Bs = Bsh[cb];
    // A 16x32 f16 frag: lanes0-15 halves0-7=K0..7, halves8-15=K16..23;
    //                   lanes16-31 shift K by +8.
    Frag16 a[2], b[4];
#pragma unroll
    for (int i = 0; i < 2; ++i) {
      int m = wm * 32 + i * 16 + ml;
      a[i].h2[0] = cvt8(As + m * ASTR + 8 * hi);
      a[i].h2[1] = cvt8(As + m * ASTR + 16 + 8 * hi);
    }
    // B 32x16 f16 frag: lanes0-15 halves0-7=K0..7, halves8-15=K16..23;
    //                   lanes16-31 shift K by +8.
#pragma unroll
    for (int j = 0; j < 4; ++j) {
      int n = wn * 64 + j * 16 + ml;
      b[j].h2[0] = cvt8(Bs + n * ASTR + 8 * hi);
      b[j].h2[1] = cvt8(Bs + n * ASTR + 16 + 8 * hi);
    }
#pragma unroll
    for (int i = 0; i < 2; ++i)
#pragma unroll
      for (int j = 0; j < 4; ++j)
        acc[i][j] = __builtin_amdgcn_wmma_f32_16x16x32_f16(
            false, a[i].v, false, b[j].v, (short)0, acc[i][j], false, false);
  };

  const int nk = K / BK;
  issue_tile();                             // prefetch depth 2
  if (nk > 1) issue_tile();

  int cb = 0;                               // rotating compute-buffer index
  for (int kt = 0; kt < nk - 1; ++kt) {     // hot loop: always partial wait
    wait_async8();                          // tile kt done; kt+1 still in flight
    __syncthreads();                        // all waves' tile-kt copies visible
    if (kt + 2 < nk) issue_tile();
    compute(cb);
    cb = (cb + 1 == NBUF) ? 0 : cb + 1;
    // single barrier per iteration: next iteration's top barrier orders
    // buffer reuse against all waves' reads.
  }
  wait_async0();                            // peeled last K-step
  __syncthreads();
  compute(cb);

  // ---- epilogue: C/D layout -> VGPR r holds (M = r + 8*hi, N = lane&15) ----
#pragma unroll
  for (int i = 0; i < 2; ++i) {
    int r0 = tile_m + wm * 32 + i * 16;
    int crow[8]; bool rok[8];
#pragma unroll
    for (int r = 0; r < 8; ++r) {
      int row = r0 + r + 8 * hi;
      rok[r]  = row < Md;
      crow[r] = rok[r] ? (mC ? mC[row] : row) : 0;
    }
#pragma unroll
    for (int j = 0; j < 4; ++j) {
      int col = tile_n + wn * 64 + j * 16 + ml;
      if (col >= N) continue;
      float bv = (epi == 1) ? bias[col] : 0.f;
#pragma unroll
      for (int r = 0; r < 8; ++r) {
        if (rok[r]) {
          float v = acc[i][j][r];
          if (epi == 1) v = softplusf(v + bv);
          C[(size_t)crow[r] * ldc + col] = v;
        }
      }
    }
  }
}

// ---------------------------------------------------------------------------
// RMSNorm: one block (256 thr) per row.
// ---------------------------------------------------------------------------
__global__ __launch_bounds__(256) void rmsnorm_kernel(
    const float* __restrict__ in, const float* __restrict__ w, float* __restrict__ out,
    int width, int in_stride, int in_off, int out_stride)
{
  int row = blockIdx.x;
  const float* p = in + (size_t)row * in_stride + in_off;
  float s = 0.f;
  for (int i = threadIdx.x; i < width; i += 256) { float v = p[i]; s += v * v; }
  s += __shfl_xor(s, 1);  s += __shfl_xor(s, 2);
  s += __shfl_xor(s, 4);  s += __shfl_xor(s, 8);  s += __shfl_xor(s, 16);
  __shared__ float red[8];
  if ((threadIdx.x & 31) == 0) red[threadIdx.x >> 5] = s;
  __syncthreads();
  float tot = 0.f;
#pragma unroll
  for (int i = 0; i < 8; ++i) tot += red[i];
  float scale = rsqrtf(tot / (float)width + 1e-6f);
  for (int i = threadIdx.x; i < width; i += 256)
    out[(size_t)row * out_stride + i] = p[i] * scale * w[i];
}

// Fused residual-add + RMSNorm: resid = a + b (stored), norm = rmsnorm(resid)*w
__global__ __launch_bounds__(256) void add_rmsnorm_kernel(
    const float* __restrict__ a, const float* __restrict__ b, const float* __restrict__ w,
    float* __restrict__ resid, float* __restrict__ normed, int width)
{
  int row = blockIdx.x;
  const float* pa = a + (size_t)row * width;
  const float* pb = b + (size_t)row * width;
  float s = 0.f;
  for (int i = threadIdx.x; i < width; i += 256) {
    float v = pa[i] + pb[i];
    resid[(size_t)row * width + i] = v;
    s += v * v;
  }
  s += __shfl_xor(s, 1);  s += __shfl_xor(s, 2);
  s += __shfl_xor(s, 4);  s += __shfl_xor(s, 8);  s += __shfl_xor(s, 16);
  __shared__ float red[8];
  if ((threadIdx.x & 31) == 0) red[threadIdx.x >> 5] = s;
  __syncthreads();
  float tot = 0.f;
#pragma unroll
  for (int i = 0; i < 8; ++i) tot += red[i];
  float scale = rsqrtf(tot / (float)width + 1e-6f);
  for (int i = threadIdx.x; i < width; i += 256)
    normed[(size_t)row * width + i] = resid[(size_t)row * width + i] * scale * w[i];
}

// ---------------------------------------------------------------------------
// Causal depthwise conv1d (K=4) + bias + SiLU.  x lives in xz[:, 0:Di].
// ---------------------------------------------------------------------------
__global__ __launch_bounds__(256) void conv_silu_kernel(
    const float* __restrict__ xz, const float* __restrict__ cw,
    const float* __restrict__ cb, float* __restrict__ xc)
{
  int idx = blockIdx.x * 256 + threadIdx.x;
  if (idx >= T_ * Di_) return;
  int t = idx / Di_, d = idx - t * Di_;
  float acc = cb[d];
#pragma unroll
  for (int k = 0; k < 4; ++k) {
    int tt = t - 3 + k;
    if (tt >= 0) acc += cw[d * 4 + k] * xz[(size_t)tt * (2 * Di_) + d];
  }
  xc[idx] = siluf(acc);
}

// ---------------------------------------------------------------------------
// Selective scan. One wave owns 2 channels x 16 states; lanes 0-15 / 16-31 are
// the two channels, lane&15 = state n. Cross-lane sum over n with shfl_xor<=8
// (stays inside each 16-lane half on wave32).
// ---------------------------------------------------------------------------
__global__ __launch_bounds__(256) void scan_kernel(
    const float* __restrict__ dtf, const float* __restrict__ xc,
    const float* __restrict__ Bn, const float* __restrict__ Cn,
    const float* __restrict__ A_log, const float* __restrict__ D_param,
    float* __restrict__ y)
{
  int gwave = (blockIdx.x * 256 + threadIdx.x) >> 5;   // 0..2047
  int lane  = threadIdx.x & 31;
  int n = lane & 15;
  int d = gwave * 2 + (lane >> 4);                     // 0..4095
  float a  = -__expf(A_log[d * N_ + n]);
  float dp = D_param[d];
  float h = 0.f;
  for (int t = 0; t < T_; ++t) {
    float dt = dtf[(size_t)t * Di_ + d];
    float xt = xc [(size_t)t * Di_ + d];
    float bt = Bn[t * N_ + n];
    float ct = Cn[t * N_ + n];
    h = h * __expf(dt * a) + dt * xt * bt;
    float c = h * ct;
    c += __shfl_xor(c, 1); c += __shfl_xor(c, 2);
    c += __shfl_xor(c, 4); c += __shfl_xor(c, 8);
    if (n == 0) y[(size_t)t * Di_ + d] = c + xt * dp;
  }
}

// y *= silu(z), z = xz[:, Di:2Di]
__global__ __launch_bounds__(256) void gate_kernel(
    float* __restrict__ y, const float* __restrict__ xz)
{
  int idx = blockIdx.x * 256 + threadIdx.x;
  if (idx >= T_ * Di_) return;
  int t = idx / Di_, d = idx - t * Di_;
  y[idx] *= siluf(xz[(size_t)t * (2 * Di_) + Di_ + d]);
}

// ---------------------------------------------------------------------------
// MoE routing: softmax over 8 logits, top-2 (first-max tie-break like top_k),
// bucket tokens per expert via atomics (order-independent result).
// ---------------------------------------------------------------------------
__global__ void init_cnt_kernel(int* cnt)
{
  if (threadIdx.x < E_) cnt[threadIdx.x] = 0;
}

__global__ __launch_bounds__(256) void route_kernel(
    const float* __restrict__ logits, float* __restrict__ gate,
    int* __restrict__ cnt, int* __restrict__ toklist, int* __restrict__ slotlist)
{
  int t = blockIdx.x * 256 + threadIdx.x;
  if (t >= T_) return;
  float l[E_], mx = -1e30f;
#pragma unroll
  for (int e = 0; e < E_; ++e) { l[e] = logits[t * E_ + e]; mx = fmaxf(mx, l[e]); }
  float s = 0.f;
#pragma unroll
  for (int e = 0; e < E_; ++e) { l[e] = __expf(l[e] - mx); s += l[e]; }
  float inv = 1.f / s;
  int i0 = 0; float v0 = -1.f;
#pragma unroll
  for (int e = 0; e < E_; ++e) { float p = l[e] * inv; if (p > v0) { v0 = p; i0 = e; } }
  int i1 = -1; float v1 = -1.f;
#pragma unroll
  for (int e = 0; e < E_; ++e) {
    if (e == i0) continue;
    float p = l[e] * inv; if (p > v1) { v1 = p; i1 = e; }
  }
  gate[2 * t]     = v0;
  gate[2 * t + 1] = v1;
  int j0 = atomicAdd(&cnt[i0], 1);
  toklist[i0 * T_ + j0]  = t;
  slotlist[i0 * T_ + j0] = 2 * t;
  int j1 = atomicAdd(&cnt[i1], 1);
  toklist[i1 * T_ + j1]  = t;
  slotlist[i1 * T_ + j1] = 2 * t + 1;
}

// h = silu(g) * u, in place into g. All 2T slots valid (exactly 2 experts/token).
__global__ __launch_bounds__(256) void swiglu_kernel(
    float* __restrict__ g, const float* __restrict__ u, long long n)
{
  long long idx = (long long)blockIdx.x * 256 + threadIdx.x;
  if (idx >= n) return;
  g[idx] = siluf(g[idx]) * u[idx];
}

// moe_out[t] = gate[2t]*yexp[2t] + gate[2t+1]*yexp[2t+1]
__global__ __launch_bounds__(256) void combine_kernel(
    const float* __restrict__ yexp, const float* __restrict__ gate,
    float* __restrict__ out)
{
  int idx = blockIdx.x * 256 + threadIdx.x;
  if (idx >= T_ * D_) return;
  int t = idx / D_, d = idx - t * D_;
  out[idx] = gate[2 * t]     * yexp[(size_t)(2 * t)     * D_ + d]
           + gate[2 * t + 1] * yexp[(size_t)(2 * t + 1) * D_ + d];
}

// ---------------------------------------------------------------------------
extern "C" void kernel_launch(void* const* d_in, const int* in_sizes, int n_in,
                              void* d_out, int out_size, void* d_ws, size_t ws_size,
                              hipStream_t stream)
{
  (void)in_sizes; (void)n_in; (void)out_size; (void)ws_size;
  const float* hidden    = (const float*)d_in[0];
  const float* ln_in_w   = (const float*)d_in[1];
  const float* in_proj_w = (const float*)d_in[2];
  const float* conv_w    = (const float*)d_in[3];
  const float* conv_b    = (const float*)d_in[4];
  const float* x_proj_w  = (const float*)d_in[5];
  const float* dt_ln_w   = (const float*)d_in[6];
  const float* b_ln_w    = (const float*)d_in[7];
  const float* c_ln_w    = (const float*)d_in[8];
  const float* dt_proj_w = (const float*)d_in[9];
  const float* dt_proj_b = (const float*)d_in[10];
  const float* A_log     = (const float*)d_in[11];
  const float* D_param   = (const float*)d_in[12];
  const float* out_proj_w= (const float*)d_in[13];
  const float* ln_ff_w   = (const float*)d_in[14];
  const float* router_w  = (const float*)d_in[15];
  const float* w1        = (const float*)d_in[16];
  const float* w3        = (const float*)d_in[17];
  const float* w2        = (const float*)d_in[18];

  char* ws = (char*)d_ws;
  size_t off = 0;
  auto alloc = [&](size_t bytes) -> void* {
    void* p = ws + off;
    off += (bytes + 255) & ~(size_t)255;
    return p;
  };
  float* hsn    = (float*)alloc((size_t)T_ * D_ * 4);
  float* xz     = (float*)alloc((size_t)T_ * 2 * Di_ * 4);
  float* xc     = (float*)alloc((size_t)T_ * Di_ * 4);
  float* xdb    = (float*)alloc((size_t)T_ * (R_ + 2 * N_) * 4);
  float* dtn    = (float*)alloc((size_t)T_ * R_ * 4);
  float* Bn     = (float*)alloc((size_t)T_ * N_ * 4);
  float* Cn     = (float*)alloc((size_t)T_ * N_ * 4);
  float* dtf    = (float*)alloc((size_t)T_ * Di_ * 4);
  float* yb     = (float*)alloc((size_t)T_ * Di_ * 4);
  float* mamba  = (float*)alloc((size_t)T_ * D_ * 4);
  float* hs2    = (float*)alloc((size_t)T_ * D_ * 4);
  float* logits = (float*)alloc((size_t)T_ * E_ * 4);
  float* gate   = (float*)alloc((size_t)2 * T_ * 4);
  int*   cnt    = (int*)  alloc((size_t)E_ * 4);
  int*   toklist= (int*)  alloc((size_t)E_ * T_ * 4);
  int*   slotlist=(int*)  alloc((size_t)E_ * T_ * 4);
  float* gbuf   = (float*)alloc((size_t)2 * T_ * I_ * 4);
  float* ubuf   = (float*)alloc((size_t)2 * T_ * I_ * 4);
  float* yexp   = (float*)alloc((size_t)2 * T_ * D_ * 4);

  float* moe_out   = (float*)d_out;                    // [T, D]
  float* resid_out = (float*)d_out + (size_t)T_ * D_;  // [T, D]

  auto nTiles = [](int n) { return (n + BN - 1) / BN; };

  // 1) hs = rmsnorm(hidden, ln_in_w)
  rmsnorm_kernel<<<T_, 256, 0, stream>>>(hidden, ln_in_w, hsn, D_, D_, 0, D_);

  // 2) xz = hs @ in_proj_w.T   [1024 x 8192]
  gemm_wmma<<<dim3(nTiles(2 * Di_), T_ / BM, 1), 256, 0, stream>>>(
      hsn, in_proj_w, xz, T_, 2 * Di_, D_, D_, 2 * Di_,
      nullptr, nullptr, nullptr, 0, nullptr, 0);

  // 3) xc = silu(conv(x) + b)
  conv_silu_kernel<<<(T_ * Di_) / 256, 256, 0, stream>>>(xz, conv_w, conv_b, xc);

  // 4) xdb = xc @ x_proj_w.T   [1024 x 160]
  gemm_wmma<<<dim3(nTiles(R_ + 2 * N_), T_ / BM, 1), 256, 0, stream>>>(
      xc, x_proj_w, xdb, T_, R_ + 2 * N_, Di_, Di_, R_ + 2 * N_,
      nullptr, nullptr, nullptr, 0, nullptr, 0);

  // 5) dt/B/C rmsnorms
  rmsnorm_kernel<<<T_, 256, 0, stream>>>(xdb, dt_ln_w, dtn, R_, R_ + 2 * N_, 0,       R_);
  rmsnorm_kernel<<<T_, 256, 0, stream>>>(xdb, b_ln_w,  Bn,  N_, R_ + 2 * N_, R_,      N_);
  rmsnorm_kernel<<<T_, 256, 0, stream>>>(xdb, c_ln_w,  Cn,  N_, R_ + 2 * N_, R_ + N_, N_);

  // 6) dtf = softplus(dtn @ dt_proj_w.T + dt_proj_b)   [1024 x 4096]
  gemm_wmma<<<dim3(nTiles(Di_), T_ / BM, 1), 256, 0, stream>>>(
      dtn, dt_proj_w, dtf, T_, Di_, R_, R_, Di_,
      nullptr, nullptr, nullptr, 0, dt_proj_b, 1);

  // 7) selective scan -> yb (+ xc * D)
  scan_kernel<<<(Di_ / 2) / 8, 256, 0, stream>>>(dtf, xc, Bn, Cn, A_log, D_param, yb);

  // 8) yb *= silu(z)
  gate_kernel<<<(T_ * Di_) / 256, 256, 0, stream>>>(yb, xz);

  // 9) mamba_out = yb @ out_proj_w.T   [1024 x 2048]
  gemm_wmma<<<dim3(nTiles(D_), T_ / BM, 1), 256, 0, stream>>>(
      yb, out_proj_w, mamba, T_, D_, Di_, Di_, D_,
      nullptr, nullptr, nullptr, 0, nullptr, 0);

  // 10) residual = mamba_out + hidden; hs2 = rmsnorm(residual, ln_ff_w)
  add_rmsnorm_kernel<<<T_, 256, 0, stream>>>(mamba, hidden, ln_ff_w, resid_out, hs2, D_);

  // 11) router logits [1024 x 8]
  gemm_wmma<<<dim3(1, T_ / BM, 1), 256, 0, stream>>>(
      hs2, router_w, logits, T_, E_, D_, D_, E_,
      nullptr, nullptr, nullptr, 0, nullptr, 0);

  // 12-13) top-2 routing + per-expert token buckets
  init_cnt_kernel<<<1, 32, 0, stream>>>(cnt);
  route_kernel<<<(T_ + 255) / 256, 256, 0, stream>>>(logits, gate, cnt, toklist, slotlist);

  // 14-15) grouped GEMMs: g/u for routed rows only (dynamic M = cnt[e])
  gemm_wmma<<<dim3(nTiles(I_), T_ / BM, E_), 256, 0, stream>>>(
      hs2, w1, gbuf, T_, I_, D_, D_, I_,
      toklist, slotlist, cnt, T_, nullptr, 0);
  gemm_wmma<<<dim3(nTiles(I_), T_ / BM, E_), 256, 0, stream>>>(
      hs2, w3, ubuf, T_, I_, D_, D_, I_,
      toklist, slotlist, cnt, T_, nullptr, 0);

  // 16) h = silu(g) * u (all 2T slots populated)
  {
    long long nsw = (long long)2 * T_ * I_;
    swiglu_kernel<<<(unsigned)((nsw + 255) / 256), 256, 0, stream>>>(gbuf, ubuf, nsw);
  }

  // 17) yexp = h @ w2.T per expert, scattered back to slots
  gemm_wmma<<<dim3(nTiles(D_), T_ / BM, E_), 256, 0, stream>>>(
      gbuf, w2, yexp, T_, D_, I_, I_, D_,
      slotlist, slotlist, cnt, T_, nullptr, 0);

  // 18) moe_out = gate-weighted sum of the two slots
  combine_kernel<<<(T_ * D_) / 256, 256, 0, stream>>>(yexp, gate, moe_out);
}
